// GaussianKernel_8581344657572
// MI455X (gfx1250) — matrix operand order repaired and checked by reading, not compiled, not run on hardware
//
#include <hip/hip_runtime.h>
#include <hip/hip_bf16.h>

typedef __attribute__((ext_vector_type(16))) __bf16 v16bf;
typedef __attribute__((ext_vector_type(8)))  __bf16 v8bf;
typedef __attribute__((ext_vector_type(8)))  float  v8f;

#define N_PTS 4096
#define DIMS  512
#define BLK   128      // output tile per 256-thread block
#define KC    32       // K chunk per LDS stage
#define LDSS  40       // LDS row stride in bf16 elems (80B: 16B aligned, bank-friendly)

// workspace layout (float offsets)
#define WS_SQ      0                       // 4096 row squared norms
#define WS_COLPART 4096                    // 64 x 512 column partial sums
#define WS_SQPART  (4096 + 64 * 512)       // 64 partial sums of ||x||^2
#define WS_SCALE   (WS_SQPART + 64)        // 1: 1/(2*sigma^2)

#define CAT16(a, b) __builtin_shufflevector((a), (b), 0,1,2,3,4,5,6,7,8,9,10,11,12,13,14,15)

// ---------------------------------------------------------------------------
// Kernel 1: per-row squared norms + per-block column sums + per-block sum(sq)
// 64 blocks x 256 threads; block b owns rows [b*64, b*64+64)
// ---------------------------------------------------------------------------
__global__ __launch_bounds__(256) void gk_stats(const float* __restrict__ X,
                                                float* __restrict__ ws) {
    const int b    = blockIdx.x;
    const int tid  = threadIdx.x;
    const int wave = tid >> 5;
    const int lane = tid & 31;
    const int r0   = b * 64;

    __shared__ float wsum[8];

    float rowacc = 0.0f;
    for (int rr = 0; rr < 8; ++rr) {
        const int row = r0 + wave * 8 + rr;
        const float* xr = X + (size_t)row * DIMS;
        float s = 0.0f;
#pragma unroll
        for (int j = 0; j < DIMS / 32; ++j) {
            float v = xr[lane + 32 * j];
            s = fmaf(v, v, s);
        }
#pragma unroll
        for (int off = 16; off >= 1; off >>= 1) s += __shfl_xor(s, off, 32);
        if (lane == 0) ws[WS_SQ + row] = s;
        rowacc += s;                       // full sum broadcast to all lanes
    }
    if (lane == 0) wsum[wave] = rowacc;
    __syncthreads();
    if (tid == 0) {
        float t = 0.0f;
        for (int w = 0; w < 8; ++w) t += wsum[w];
        ws[WS_SQPART + b] = t;
    }

    // column partial sums over this block's 64 rows (coalesced across threads)
    for (int c = tid; c < DIMS; c += 256) {
        float s = 0.0f;
        for (int r = 0; r < 64; ++r) s += X[(size_t)(r0 + r) * DIMS + c];
        ws[WS_COLPART + b * DIMS + c] = s;
    }
}

// ---------------------------------------------------------------------------
// Kernel 2: sigma^2 = 2*mean(||x||^2) - 2*||mean(X)||^2 ; store 1/(2 sigma^2)
// 1 block x 512 threads, fully deterministic reduction order
// ---------------------------------------------------------------------------
__global__ __launch_bounds__(512) void gk_finalize(float* __restrict__ ws) {
    const int tid = threadIdx.x;
    __shared__ float red[512];

    float musq = 0.0f;
    if (tid < DIMS) {
        float s = 0.0f;
        for (int p = 0; p < 64; ++p) s += ws[WS_COLPART + p * DIMS + tid];
        float mu = s * (1.0f / (float)N_PTS);
        musq = mu * mu;
    }
    red[tid] = musq;
    __syncthreads();
    for (int off = 256; off > 0; off >>= 1) {
        if (tid < off) red[tid] += red[tid + off];
        __syncthreads();
    }
    const float sum_musq = red[0];
    __syncthreads();

    red[tid] = (tid < 64) ? ws[WS_SQPART + tid] : 0.0f;
    __syncthreads();
    for (int off = 256; off > 0; off >>= 1) {
        if (tid < off) red[tid] += red[tid + off];
        __syncthreads();
    }
    if (tid == 0) {
        const float mean_sq = red[0] * (1.0f / (float)N_PTS);
        const float sigma2  = 1.0f * (2.0f * mean_sq - 2.0f * sum_musq); // ALPHA=1
        ws[WS_SCALE] = 1.0f / (2.0f * sigma2);
    }
}

// ---------------------------------------------------------------------------
// Kernel 3: fused WMMA GEMM (bf16x3 split for ~fp32 accuracy) + exp epilogue
// grid (32,32), 256 threads (8 waves); each block -> 128x128 output tile.
// Wave grid 4x2: each wave owns 32 rows x 64 cols = 2x4 tiles of 16x16.
// ---------------------------------------------------------------------------
__global__ __launch_bounds__(256) void gk_gemm_exp(const float* __restrict__ X,
                                                   const float* __restrict__ ws,
                                                   float* __restrict__ out) {
    __shared__ __attribute__((aligned(16))) __bf16 sAh[BLK * LDSS];
    __shared__ __attribute__((aligned(16))) __bf16 sAl[BLK * LDSS];
    __shared__ __attribute__((aligned(16))) __bf16 sBh[BLK * LDSS];
    __shared__ __attribute__((aligned(16))) __bf16 sBl[BLK * LDSS];
    __shared__ float sSqM[BLK];
    __shared__ float sSqN[BLK];

    const int tid  = threadIdx.x;
    const int lane = tid & 31;
    const int wave = tid >> 5;
    const int wm   = wave >> 1;   // 0..3: 32-row strip
    const int wn   = wave & 1;    // 0..1: 64-col strip
    const int m0   = blockIdx.y * BLK;
    const int n0   = blockIdx.x * BLK;

    if (tid < 128) sSqM[tid]       = ws[WS_SQ + m0 + tid];
    else           sSqN[tid - 128] = ws[WS_SQ + n0 + tid - 128];
    const float scale = ws[WS_SCALE];

    v8f acc[2][4] = {};

    const int rsel = lane & 15;
    const int hsel = lane >> 4;   // half-wave select per ISA fragment layout

    for (int kc = 0; kc < DIMS; kc += KC) {
        __syncthreads();   // previous consumers done (also covers sSq* stores)

        // stage X[m0..][kc..] and X[n0..][kc..] into LDS as bf16 hi/lo split
#pragma unroll
        for (int i = 0; i < 4; ++i) {
            const int idx = tid + i * 256;   // 0..1023
            const int r   = idx >> 3;        // 0..127
            const int q   = idx & 7;         // float4 within 32-col chunk
            const float4 va = *(const float4*)(X + (size_t)(m0 + r) * DIMS + kc + q * 4);
            const float4 vb = *(const float4*)(X + (size_t)(n0 + r) * DIMS + kc + q * 4);
            const int o = r * LDSS + q * 4;
            {
                float x; __bf16 h;
                x = va.x; h = (__bf16)x; sAh[o+0] = h; sAl[o+0] = (__bf16)(x - (float)h);
                x = va.y; h = (__bf16)x; sAh[o+1] = h; sAl[o+1] = (__bf16)(x - (float)h);
                x = va.z; h = (__bf16)x; sAh[o+2] = h; sAl[o+2] = (__bf16)(x - (float)h);
                x = va.w; h = (__bf16)x; sAh[o+3] = h; sAl[o+3] = (__bf16)(x - (float)h);
                x = vb.x; h = (__bf16)x; sBh[o+0] = h; sBl[o+0] = (__bf16)(x - (float)h);
                x = vb.y; h = (__bf16)x; sBh[o+1] = h; sBl[o+1] = (__bf16)(x - (float)h);
                x = vb.z; h = (__bf16)x; sBh[o+2] = h; sBl[o+2] = (__bf16)(x - (float)h);
                x = vb.w; h = (__bf16)x; sBh[o+3] = h; sBl[o+3] = (__bf16)(x - (float)h);
            }
        }
        __syncthreads();

        // A fragments: 16x32 bf16, lane L: M=L&15, K = 8*(L>>4)+{0..7, 16..23}
        v16bf afh[2], afl[2];
        const int koff = hsel * 8;
#pragma unroll
        for (int mt = 0; mt < 2; ++mt) {
            const int row = wm * 32 + mt * 16 + rsel;
            const __bf16* ph = &sAh[row * LDSS];
            const __bf16* pl = &sAl[row * LDSS];
            v8bf x0 = *(const v8bf*)(ph + koff);
            v8bf x1 = *(const v8bf*)(ph + 16 + koff);
            afh[mt] = CAT16(x0, x1);
            v8bf y0 = *(const v8bf*)(pl + koff);
            v8bf y1 = *(const v8bf*)(pl + 16 + koff);
            afl[mt] = CAT16(y0, y1);
        }

        // B fragments: 32x16 bf16, lane L: N=L&15, K = 8*(L>>4)+{0..7, 16..23}
        const int kb = hsel * 8;
#pragma unroll
        for (int nt = 0; nt < 4; ++nt) {
            const int rowb = wn * 64 + nt * 16 + rsel;
            const __bf16* ph = &sBh[rowb * LDSS + kb];
            const __bf16* pl = &sBl[rowb * LDSS + kb];
            v8bf b0 = *(const v8bf*)(ph);
            v8bf b1 = *(const v8bf*)(ph + 16);
            const v16bf bfh = CAT16(b0, b1);
            v8bf c0 = *(const v8bf*)(pl);
            v8bf c1 = *(const v8bf*)(pl + 16);
            const v16bf bfl = CAT16(c0, c1);
#pragma unroll
            for (int mt = 0; mt < 2; ++mt) {
                // bf16x3: hi*hi + hi*lo + lo*hi  (lo*lo ~2^-18, negligible)
                acc[mt][nt] = __builtin_amdgcn_wmma_f32_16x16x32_bf16(
                    false, afh[mt], false, bfh, (short)0, acc[mt][nt], false, false);
                acc[mt][nt] = __builtin_amdgcn_wmma_f32_16x16x32_bf16(
                    false, afh[mt], false, bfl, (short)0, acc[mt][nt], false, false);
                acc[mt][nt] = __builtin_amdgcn_wmma_f32_16x16x32_bf16(
                    false, afl[mt], false, bfh, (short)0, acc[mt][nt], false, false);
            }
        }
    }

    // epilogue: out[m][n] = exp(-(sq_m + sq_n - 2 g) * 1/(2 sigma^2))
    // C/D layout: VGPR v, lane L -> M = 8*(L>>4)+v, N = L&15
#pragma unroll
    for (int mt = 0; mt < 2; ++mt) {
        const int mloc0 = wm * 32 + mt * 16 + hsel * 8;
#pragma unroll
        for (int nt = 0; nt < 4; ++nt) {
            const int nloc = wn * 64 + nt * 16 + rsel;
            const float sqn = sSqN[nloc];
            const int n = n0 + nloc;
#pragma unroll
            for (int v = 0; v < 8; ++v) {
                const int mloc = mloc0 + v;
                const float g  = acc[mt][nt][v];
                const float d2 = sSqM[mloc] + sqn - 2.0f * g;
                out[(size_t)(m0 + mloc) * N_PTS + n] = __expf(-d2 * scale);
            }
        }
    }
}

// ---------------------------------------------------------------------------
extern "C" void kernel_launch(void* const* d_in, const int* in_sizes, int n_in,
                              void* d_out, int out_size, void* d_ws, size_t ws_size,
                              hipStream_t stream) {
    (void)in_sizes; (void)n_in; (void)out_size; (void)ws_size;
    const float* X  = (const float*)d_in[0];
    float*       ws = (float*)d_ws;
    float*       o  = (float*)d_out;

    gk_stats<<<64, 256, 0, stream>>>(X, ws);
    gk_finalize<<<1, 512, 0, stream>>>(ws);
    dim3 grid(N_PTS / BLK, N_PTS / BLK);   // 32 x 32
    gk_gemm_exp<<<grid, 256, 0, stream>>>(X, ws, o);
}
